// VideoDepthAnything_71957882077583
// MI455X (gfx1250) — matrix operand order repaired and checked by reading, not compiled, not run on hardware
//
#include <hip/hip_runtime.h>
#include <math.h>

// ---------------- problem constants (match reference) ----------------
#define B_   2
#define T_   8
#define Hh   37
#define Ww   37
#define N_   (Hh * Ww)      // 1369
#define C_   384
#define NH   8
#define MP   8
#define HD   (C_ / NH)      // 48
#define NOFF (NH * MP * 2)  // 128
#define KSTEPS (C_ / 32)    // 12

typedef __attribute__((ext_vector_type(16))) __bf16 v16bf;
typedef __attribute__((ext_vector_type(8)))  float  v8f;
typedef __attribute__((ext_vector_type(4)))  int    v4i;

// =====================================================================
// Fragment packing. WMMA fragments are stored as [tile][k32][lane][16]
// bf16 (32 bytes per lane, contiguous), matching CDNA5 ISA 7.12.2:
//   A (16x32): lane = row (lane&15); lo-half VGPRs K = 8*halfSel + 2vi,
//              hi-half VGPRs K = 16 + 8*halfSel + 2(vi-4)
//   B (32x16): lane = col (lane&15); same K mapping as A
// =====================================================================
__global__ __launch_bounds__(256)
void vda_pack_a_bf16(const float* __restrict__ A, __bf16* __restrict__ out,
                     int M, int K)
{
    const int gid = blockIdx.x * blockDim.x + threadIdx.x;
    const int kSteps = K >> 5;
    const int total = (M >> 4) * kSteps * 32;
    if (gid >= total) return;

    const int lane    = gid & 31;
    const int frag    = gid >> 5;
    const int kk32    = frag % kSteps;
    const int mt      = frag / kSteps;
    const int halfSel = lane >> 4;
    const int lrow    = lane & 15;

    const float* Ap = A + (size_t)(mt * 16 + lrow) * K;
    v16bf v;
    #pragma unroll
    for (int vi = 0; vi < 8; ++vi) {
        const int k0 = (vi < 4) ? (kk32 * 32 + halfSel * 8 + 2 * vi)
                                : (kk32 * 32 + 16 + halfSel * 8 + 2 * (vi - 4));
        float2 av = *(const float2*)(Ap + k0);
        v[2 * vi]     = (__bf16)av.x;
        v[2 * vi + 1] = (__bf16)av.y;
    }
    ((v16bf*)out)[gid] = v;
}

__global__ __launch_bounds__(256)
void vda_pack_b_bf16(const float* __restrict__ W, __bf16* __restrict__ out,
                     int K, int Ncols)
{
    const int gid = blockIdx.x * blockDim.x + threadIdx.x;
    const int kSteps = K >> 5;
    const int total = (Ncols >> 4) * kSteps * 32;
    if (gid >= total) return;

    const int lane    = gid & 31;
    const int frag    = gid >> 5;
    const int kk32    = frag % kSteps;
    const int nt      = frag / kSteps;
    const int halfSel = lane >> 4;
    const int col     = nt * 16 + (lane & 15);

    v16bf v;
    #pragma unroll
    for (int vi = 0; vi < 8; ++vi) {
        const int k0 = (vi < 4) ? (kk32 * 32 + halfSel * 8 + 2 * vi)
                                : (kk32 * 32 + 16 + halfSel * 8 + 2 * (vi - 4));
        v[2 * vi]     = (__bf16)W[(size_t)k0       * Ncols + col];
        v[2 * vi + 1] = (__bf16)W[(size_t)(k0 + 1) * Ncols + col];
    }
    ((v16bf*)out)[gid] = v;
}

// =====================================================================
// Packed WMMA GEMM with LDS-staged B.
// Block = 256 threads (8 waves) owns 8 M-tiles x one 4-tile N-group.
// The N-group's packed B is one contiguous 48KB slice -> staged into LDS
// once per block (async-to-LDS when available), then each wave runs a
// fully unrolled K loop: ds_load B fragments + pipelined global A
// fragments + 4 WMMAs per K-step.
// =====================================================================
__global__ __launch_bounds__(256)
void vda_gemm_packed(const __bf16* __restrict__ Apack,
                     const __bf16* __restrict__ Bpack,
                     const float* __restrict__ bias,
                     float* __restrict__ Cout,
                     int Mtiles, int Ncols)
{
    const int ngroups = Ncols >> 6;                 // groups of 4 N-tiles
    const int ng      = blockIdx.x % ngroups;
    const int mtBase  = (blockIdx.x / ngroups) * 8;
    const int waveId  = threadIdx.x >> 5;
    const int lane    = threadIdx.x & 31;

    __shared__ v16bf sB[4 * KSTEPS * 32];           // 48 KB

    // ---- stage this N-group's packed B (contiguous) into LDS ----
    {
        const v16bf* src = (const v16bf*)Bpack + (size_t)ng * 4 * KSTEPS * 32;
#if __has_builtin(__builtin_amdgcn_global_load_async_to_lds_b128) && \
    __has_builtin(__builtin_amdgcn_s_wait_asynccnt)
        typedef __attribute__((address_space(1))) v4i* gv4p;
        typedef __attribute__((address_space(3))) v4i* lv4p;
        const int chunks = 4 * KSTEPS * 32 * 2;     // 16B chunks = 3072
        for (int c = threadIdx.x; c < chunks; c += 256) {
            gv4p g = (gv4p)(v4i*)((char*)src + (size_t)c * 16);
            lv4p l = (lv4p)(v4i*)((char*)sB  + (size_t)c * 16);
            __builtin_amdgcn_global_load_async_to_lds_b128(g, l, 0, 0);
        }
        __builtin_amdgcn_s_wait_asynccnt(0);
#else
        for (int c = threadIdx.x; c < 4 * KSTEPS * 32; c += 256)
            sB[c] = src[c];
#endif
    }
    __syncthreads();

    const int mt = mtBase + waveId;                 // wave-uniform guard
    if (mt < Mtiles) {
        const v16bf* Af = (const v16bf*)Apack + (size_t)mt * KSTEPS * 32 + lane;

        v8f acc0 = {}, acc1 = {}, acc2 = {}, acc3 = {};
        v16bf a_cur = Af[0];

        #pragma unroll
        for (int k = 0; k < KSTEPS; ++k) {
            v16bf a_nxt = a_cur;
            if (k + 1 < KSTEPS) a_nxt = Af[(size_t)(k + 1) * 32];   // pipeline A

            const v16bf b0 = sB[(0 * KSTEPS + k) * 32 + lane];
            const v16bf b1 = sB[(1 * KSTEPS + k) * 32 + lane];
            const v16bf b2 = sB[(2 * KSTEPS + k) * 32 + lane];
            const v16bf b3 = sB[(3 * KSTEPS + k) * 32 + lane];

            acc0 = __builtin_amdgcn_wmma_f32_16x16x32_bf16(false, a_cur, false, b0, (short)0, acc0, false, false);
            acc1 = __builtin_amdgcn_wmma_f32_16x16x32_bf16(false, a_cur, false, b1, (short)0, acc1, false, false);
            acc2 = __builtin_amdgcn_wmma_f32_16x16x32_bf16(false, a_cur, false, b2, (short)0, acc2, false, false);
            acc3 = __builtin_amdgcn_wmma_f32_16x16x32_bf16(false, a_cur, false, b3, (short)0, acc3, false, false);

            a_cur = a_nxt;
        }

        // D layout: VGPR i -> row (mt*16 + i + 8*halfSel), col = tile_n*16 + (lane&15)
        const int halfSel = lane >> 4;
        const int lrow    = lane & 15;
        #pragma unroll
        for (int j = 0; j < 4; ++j) {
            const v8f acc = (j == 0) ? acc0 : (j == 1) ? acc1 : (j == 2) ? acc2 : acc3;
            const int col = (ng * 4 + j) * 16 + lrow;
            const float bv = bias[col];
            #pragma unroll
            for (int i = 0; i < 8; ++i) {
                const int row = mt * 16 + i + 8 * halfSel;
                Cout[(size_t)row * Ncols + col] = acc[i] + bv;
            }
        }
    }
}

// =====================================================================
// Deformable attention core. One thread = one (b, t, n, h).
// Tap indices/weights depend only on (t,n,h,p) -> computed once, reused
// across all T reference frames. Two passes over the gathers per ref
// (logits, then weighted sum) to keep s[P][HD] out of registers.
// =====================================================================
__global__ __launch_bounds__(256)
void vda_attend(const float* __restrict__ f_bt,
                const float* __restrict__ q_ws,
                const float* __restrict__ off_ws,
                float* __restrict__ ctx)
{
    const int gid = blockIdx.x * blockDim.x + threadIdx.x;
    const int total = B_ * T_ * N_ * NH;
    if (gid >= total) return;

    const int h = gid % NH;
    const int n = (gid / NH) % N_;
    const int t = (gid / (NH * N_)) % T_;
    const int b = gid / (NH * N_ * T_);

    const float rowc  = (float)(n / Ww);
    const float colc  = (float)(n % Ww);
    const float scale = 0.14433756729740643f;   // 48^-0.5

    float4 qv[HD / 4];
    const float* qp = q_ws + ((size_t)((b * T_ + t) * N_) + n) * C_ + h * HD;
    #pragma unroll
    for (int i = 0; i < HD / 4; ++i) qv[i] = ((const float4*)qp)[i];

    const float* offp = off_ws + ((size_t)((b * T_ + t) * N_) + n) * NOFF + h * MP * 2;
    int   i00[MP], i01[MP], i10[MP], i11[MP];
    float w00[MP], w01[MP], w10[MP], w11[MP];
    #pragma unroll
    for (int p = 0; p < MP; ++p) {
        const float y  = rowc + offp[p * 2 + 0];
        const float x  = colc + offp[p * 2 + 1];
        const float y0 = floorf(y), x0 = floorf(x);
        const float wy = y - y0,    wx = x - x0;
        const int yi = (int)y0, xi = (int)x0;

        const bool vy0 = (yi     >= 0) && (yi     < Hh);
        const bool vy1 = (yi + 1 >= 0) && (yi + 1 < Hh);
        const bool vx0 = (xi     >= 0) && (xi     < Ww);
        const bool vx1 = (xi + 1 >= 0) && (xi + 1 < Ww);

        const int yc0 = min(max(yi,     0), Hh - 1);
        const int yc1 = min(max(yi + 1, 0), Hh - 1);
        const int xc0 = min(max(xi,     0), Ww - 1);
        const int xc1 = min(max(xi + 1, 0), Ww - 1);

        i00[p] = yc0 * Ww + xc0;  i01[p] = yc0 * Ww + xc1;
        i10[p] = yc1 * Ww + xc0;  i11[p] = yc1 * Ww + xc1;
        w00[p] = (vy0 && vx0) ? (1.f - wy) * (1.f - wx) : 0.f;
        w01[p] = (vy0 && vx1) ? (1.f - wy) * wx         : 0.f;
        w10[p] = (vy1 && vx0) ? wy * (1.f - wx)         : 0.f;
        w11[p] = (vy1 && vx1) ? wy * wx                 : 0.f;
    }

    float accv[HD];
    #pragma unroll
    for (int i = 0; i < HD; ++i) accv[i] = 0.f;

    for (int r = 0; r < T_; ++r) {
        const int dtr = (t > r) ? (t - r) : (r - t);
        int np = (MP * (T_ - 1 - dtr)) / (T_ - 1);
        if (np < 1) np = 1;

        const float* refbase = f_bt + (size_t)((b * T_ + r) * N_) * C_ + h * HD;

        float logit[MP];
        #pragma unroll
        for (int p = 0; p < MP; ++p) {
            if (p < np) {
                const float4* p00 = (const float4*)(refbase + (size_t)i00[p] * C_);
                const float4* p01 = (const float4*)(refbase + (size_t)i01[p] * C_);
                const float4* p10 = (const float4*)(refbase + (size_t)i10[p] * C_);
                const float4* p11 = (const float4*)(refbase + (size_t)i11[p] * C_);
                float dot = 0.f;
                #pragma unroll
                for (int i = 0; i < HD / 4; ++i) {
                    const float4 a0 = p00[i], a1 = p01[i], a2 = p10[i], a3 = p11[i];
                    const float sx = w00[p]*a0.x + w01[p]*a1.x + w10[p]*a2.x + w11[p]*a3.x;
                    const float sy = w00[p]*a0.y + w01[p]*a1.y + w10[p]*a2.y + w11[p]*a3.y;
                    const float sz = w00[p]*a0.z + w01[p]*a1.z + w10[p]*a2.z + w11[p]*a3.z;
                    const float sw = w00[p]*a0.w + w01[p]*a1.w + w10[p]*a2.w + w11[p]*a3.w;
                    dot += qv[i].x*sx + qv[i].y*sy + qv[i].z*sz + qv[i].w*sw;
                }
                logit[p] = dot * scale;
            } else {
                logit[p] = -1e30f;
            }
        }

        float mx = logit[0];
        #pragma unroll
        for (int p = 1; p < MP; ++p) mx = fmaxf(mx, logit[p]);
        float e[MP], den = 0.f;
        #pragma unroll
        for (int p = 0; p < MP; ++p) { e[p] = __expf(logit[p] - mx); den += e[p]; }
        const float inv = 1.f / den;

        #pragma unroll
        for (int p = 0; p < MP; ++p) {
            if (p < np) {
                const float ap = e[p] * inv;
                const float4* p00 = (const float4*)(refbase + (size_t)i00[p] * C_);
                const float4* p01 = (const float4*)(refbase + (size_t)i01[p] * C_);
                const float4* p10 = (const float4*)(refbase + (size_t)i10[p] * C_);
                const float4* p11 = (const float4*)(refbase + (size_t)i11[p] * C_);
                #pragma unroll
                for (int i = 0; i < HD / 4; ++i) {
                    const float4 a0 = p00[i], a1 = p01[i], a2 = p10[i], a3 = p11[i];
                    accv[4*i+0] += ap * (w00[p]*a0.x + w01[p]*a1.x + w10[p]*a2.x + w11[p]*a3.x);
                    accv[4*i+1] += ap * (w00[p]*a0.y + w01[p]*a1.y + w10[p]*a2.y + w11[p]*a3.y);
                    accv[4*i+2] += ap * (w00[p]*a0.z + w01[p]*a1.z + w10[p]*a2.z + w11[p]*a3.z);
                    accv[4*i+3] += ap * (w00[p]*a0.w + w01[p]*a1.w + w10[p]*a2.w + w11[p]*a3.w);
                }
            }
        }
    }

    const float invT = 1.0f / (float)T_;
    float* op = ctx + ((size_t)((b * T_ + t) * N_) + n) * C_ + h * HD;
    #pragma unroll
    for (int i = 0; i < HD; ++i) op[i] = accv[i] * invT;
}

// =====================================================================
extern "C" void kernel_launch(void* const* d_in, const int* in_sizes, int n_in,
                              void* d_out, int out_size, void* d_ws, size_t ws_size,
                              hipStream_t stream)
{
    const float* f_bt = (const float*)d_in[0];
    const float* Wq   = (const float*)d_in[1];
    const float* bq   = (const float*)d_in[2];
    const float* Woff = (const float*)d_in[3];
    const float* boff = (const float*)d_in[4];
    const float* Wout = (const float*)d_in[5];
    const float* bout = (const float*)d_in[6];
    float* out = (float*)d_out;

    const int M = B_ * T_ * N_;          // 21904 (divisible by 16)
    const int K = C_;                    // 384 (divisible by 32)
    const int Mtiles = M / 16;           // 1369

    // ---- workspace layout (ctx aliases the dead f_bt A-pack region) ----
    unsigned char* ws = (unsigned char*)d_ws;
    float* q_ws   = (float*)ws;                              // M*C_ f32
    float* off_ws = q_ws + (size_t)M * C_;                   // M*NOFF f32
    unsigned char* r2 = (unsigned char*)(off_ws + (size_t)M * NOFF);
    __bf16* apack = (__bf16*)r2;                             // M*K bf16 (dead after GEMMs 1-2)
    float*  ctx   = (float*)r2;                              // M*C_ f32 (written by attend)
    unsigned char* r3 = (unsigned char*)(ctx + (size_t)M * C_);
    __bf16* cpack  = (__bf16*)r3;                            // M*K bf16
    __bf16* bq_p   = (__bf16*)(r3 + (size_t)M * K * sizeof(__bf16));
    __bf16* boff_p = bq_p   + (size_t)K * C_;
    __bf16* bout_p = boff_p + (size_t)K * NOFF;

    // ---- 0) pack weights + input activations to bf16 WMMA fragments ----
    {
        int tb = (C_ / 16) * KSTEPS * 32;      // Wq / Wout
        vda_pack_b_bf16<<<(tb + 255) / 256, 256, 0, stream>>>(Wq,   bq_p,   K, C_);
        vda_pack_b_bf16<<<(tb + 255) / 256, 256, 0, stream>>>(Wout, bout_p, K, C_);
        int to = (NOFF / 16) * KSTEPS * 32;
        vda_pack_b_bf16<<<(to + 255) / 256, 256, 0, stream>>>(Woff, boff_p, K, NOFF);
        int ta = Mtiles * KSTEPS * 32;
        vda_pack_a_bf16<<<(ta + 255) / 256, 256, 0, stream>>>(f_bt, apack, M, K);
    }
    // ---- 1) q = f_bt @ Wq + bq   [M x 384] ----
    {
        const int blocks = ((Mtiles + 7) / 8) * (C_ / 64);
        vda_gemm_packed<<<blocks, 256, 0, stream>>>(apack, bq_p, bq, q_ws, Mtiles, C_);
    }
    // ---- 2) off = f_bt @ Woff + boff   [M x 128] ----
    {
        const int blocks = ((Mtiles + 7) / 8) * (NOFF / 64);
        vda_gemm_packed<<<blocks, 256, 0, stream>>>(apack, boff_p, boff, off_ws, Mtiles, NOFF);
    }
    // ---- 3) deformable attention core -> ctx [M x 384] ----
    {
        const int total = B_ * T_ * N_ * NH;
        vda_attend<<<(total + 255) / 256, 256, 0, stream>>>(f_bt, q_ws, off_ws, ctx);
    }
    // ---- 4) pack ctx, then out = ctx @ Wout + bout   [M x 384] ----
    {
        int ta = Mtiles * KSTEPS * 32;
        vda_pack_a_bf16<<<(ta + 255) / 256, 256, 0, stream>>>(ctx, cpack, M, K);
        const int blocks = ((Mtiles + 7) / 8) * (C_ / 64);
        vda_gemm_packed<<<blocks, 256, 0, stream>>>(cpack, bout_p, bout, out, Mtiles, C_);
    }
}
